// DeformableLocalCrossAttention_14826227106562
// MI455X (gfx1250) — matrix operand order repaired and checked by reading, not compiled, not run on hardware
//
#include <hip/hip_runtime.h>

#define DEVI __device__ __forceinline__

typedef __attribute__((ext_vector_type(16))) _Float16 v16h;
typedef __attribute__((ext_vector_type(8)))  float    v8f;

constexpr int B_  = 4, N_ = 1024, C_ = 384, H_ = 6, G_ = 2, K_ = 10;
constexpr int GD_ = C_ / G_;          // 192
constexpr int HD_ = C_ / H_;          // 64
constexpr float SCALE_  = 0.125f;     // HD^-0.5
constexpr float LN_EPS_ = 1e-5f;
constexpr int MQ_ = B_ * N_;               // 4096 rows (q-side GEMMs)
constexpr int M1_ = B_ * G_ * N_ * K_;     // 81920 rows (W1 GEMM)
constexpr int MI_ = B_ * N_ * K_;          // 40960 rows (interp)
constexpr int CC_ = C_ * C_;               // 147456

// ---------------------------------------------------------------------------
// WMMA fragment loaders (CDNA5 16x16x32 f16 layouts, CDNA5 ISA)
// A (16x32, MxK) row-major: per lane two contiguous 8-elem runs -> 2x b128.
// ---------------------------------------------------------------------------
DEVI v16h load_a16(const _Float16* __restrict__ A, int lda, int row, int kb, int half) {
  const _Float16* p = A + (size_t)row * lda + kb + 8 * half;
  v16h a;
#pragma unroll
  for (int v = 0; v < 4; ++v) {          // K = 8*half + {0..7}
    a[2 * v]     = p[2 * v];
    a[2 * v + 1] = p[2 * v + 1];
  }
#pragma unroll
  for (int v = 4; v < 8; ++v) {          // K = 16 + 8*half + {0..7}
    a[2 * v]     = p[16 + 2 * (v - 4)];
    a[2 * v + 1] = p[16 + 2 * (v - 4) + 1];
  }
  return a;
}
// B (32x16, KxN) from TRANSPOSED storage BT[col][k] (col-major weight):
// per lane two contiguous 8-elem runs (K = 8*half+{0..7}, 16+8*half+{0..7}) -> 2x b128.
DEVI v16h load_bT(const _Float16* __restrict__ BT, int kb, int col, int half) {
  const _Float16* p = BT + (size_t)col * C_ + kb + 8 * half;
  v16h b;
#pragma unroll
  for (int v = 0; v < 16; ++v) b[v] = p[v < 8 ? v : 16 + (v - 8)];
  return b;
}

// ---------------------------------------------------------------------------
// Kernel 1a: f32 -> f16 convert (row-major, activations)
// ---------------------------------------------------------------------------
__global__ void cvt_f16_kernel(const float* __restrict__ s, _Float16* __restrict__ d, int n) {
  int i = blockIdx.x * blockDim.x + threadIdx.x;
  if (i < n) d[i] = (_Float16)s[i];
}
// Kernel 1b: f32 -> f16 transposed convert for C_ x C_ weights: d[c][k]=s[k][c]
__global__ void cvt_f16_t_kernel(const float* __restrict__ s, _Float16* __restrict__ d) {
  int i = blockIdx.x * blockDim.x + threadIdx.x;
  if (i >= CC_) return;
  int k = i / C_, c = i % C_;
  d[(size_t)c * C_ + k] = (_Float16)s[i];
}

// ---------------------------------------------------------------------------
// Kernel 2: brute-force kNN top-K=10 over q_pos (thread per query)
// ---------------------------------------------------------------------------
__global__ void knn_kernel(const float* __restrict__ q_pos, int* __restrict__ idx) {
  int t = blockIdx.x * blockDim.x + threadIdx.x;
  if (t >= B_ * N_) return;
  int b = t / N_, n = t % N_;
  const float* base = q_pos + (size_t)b * N_ * 3;
  float px = base[n * 3 + 0], py = base[n * 3 + 1], pz = base[n * 3 + 2];
  float bd[K_]; int bi[K_];
#pragma unroll
  for (int i = 0; i < K_; ++i) { bd[i] = 3.4e38f; bi[i] = 0; }
  for (int j = 0; j < N_; ++j) {
    float dx = base[j * 3 + 0] - px, dy = base[j * 3 + 1] - py, dz = base[j * 3 + 2] - pz;
    float d = dx * dx + dy * dy + dz * dz;
    if (d < bd[K_ - 1]) {
      bd[K_ - 1] = d; bi[K_ - 1] = j;
#pragma unroll
      for (int ti = K_ - 1; ti > 0; --ti) {
        if (bd[ti] < bd[ti - 1]) {
          float td = bd[ti]; bd[ti] = bd[ti - 1]; bd[ti - 1] = td;
          int   tt = bi[ti]; bi[ti] = bi[ti - 1]; bi[ti - 1] = tt;
        }
      }
    }
  }
#pragma unroll
  for (int i = 0; i < K_; ++i) idx[t * K_ + i] = bi[i];
}

// ---------------------------------------------------------------------------
// Kernel 3: WMMA GEMM  Cf(MxC_) = A(MxC_) @ B(C_xC_) [+ bias], B transposed.
// wave = 16x64 strip (4 tiles): A frag reused by 4 WMMAs per k-step.
// ---------------------------------------------------------------------------
__global__ __launch_bounds__(128) void wmma_gemm_kernel(
    const _Float16* __restrict__ A, const _Float16* __restrict__ BT,
    const float* __restrict__ bias, float* __restrict__ Cf, int M) {
  int wid  = (blockIdx.x * blockDim.x + threadIdx.x) >> 5;
  int lane = threadIdx.x & 31;
  const int groupsN = C_ / 64;               // 6
  int rt = wid / groupsN, cg = wid % groupsN;
  if (rt * 16 >= M) return;
  int half = lane >> 4, r = lane & 15;
  v8f acc0 = {}, acc1 = {}, acc2 = {}, acc3 = {};
  for (int kb = 0; kb < C_; kb += 32) {
    __builtin_prefetch(A + (size_t)(rt * 16 + r) * C_ + kb + 32, 0, 3);
    v16h a  = load_a16(A, C_, rt * 16 + r, kb, half);
    v16h b0 = load_bT(BT, kb, cg * 64 + 0 * 16 + r, half);
    v16h b1 = load_bT(BT, kb, cg * 64 + 1 * 16 + r, half);
    v16h b2 = load_bT(BT, kb, cg * 64 + 2 * 16 + r, half);
    v16h b3 = load_bT(BT, kb, cg * 64 + 3 * 16 + r, half);
    acc0 = __builtin_amdgcn_wmma_f32_16x16x32_f16(false, a, false, b0, (short)0, acc0, false, false);
    acc1 = __builtin_amdgcn_wmma_f32_16x16x32_f16(false, a, false, b1, (short)0, acc1, false, false);
    acc2 = __builtin_amdgcn_wmma_f32_16x16x32_f16(false, a, false, b2, (short)0, acc2, false, false);
    acc3 = __builtin_amdgcn_wmma_f32_16x16x32_f16(false, a, false, b3, (short)0, acc3, false, false);
  }
  float bb[4];
#pragma unroll
  for (int j = 0; j < 4; ++j) bb[j] = bias ? bias[cg * 64 + j * 16 + r] : 0.f;
  v8f* accs[4] = { &acc0, &acc1, &acc2, &acc3 };
#pragma unroll
  for (int j = 0; j < 4; ++j) {
    int col = cg * 64 + j * 16 + r;
#pragma unroll
    for (int v = 0; v < 8; ++v) {
      int row = rt * 16 + 8 * half + v;
      Cf[(size_t)row * C_ + col] = (*accs[j])[v] + bb[j];
    }
  }
}

// ---------------------------------------------------------------------------
// Kernel 4: build shift_feat = concat(gathered v_off, broadcast group_q) as f16
// ---------------------------------------------------------------------------
__global__ void build_sf_kernel(const float* __restrict__ qp, const float* __restrict__ voff,
                                const int* __restrict__ idx, _Float16* __restrict__ sf16) {
  long long e = (long long)blockIdx.x * blockDim.x + threadIdx.x;
  if (e >= (long long)M1_ * C_) return;
  int c = (int)(e % C_);
  int m = (int)(e / C_);
  int b = m / (G_ * N_ * K_);
  int g = (m / (N_ * K_)) % G_;
  int n = (m / K_) % N_;
  int k = m % K_;
  float v;
  if (c < GD_) {
    int nb = idx[(b * N_ + n) * K_ + k];
    v = voff[(size_t)(b * N_ + nb) * C_ + g * GD_ + c];
  } else {
    v = qp[(size_t)(b * N_ + n) * C_ + g * GD_ + (c - GD_)];
  }
  sf16[e] = (_Float16)v;
}

// ---------------------------------------------------------------------------
// Kernel 5: fused  h = LN(sf@W1 + b1); gelu; off = tanh(h@W2);
//           shift_pos = q_pos[idx] + off        (wave = 16-row block)
// ---------------------------------------------------------------------------
__global__ __launch_bounds__(64) void w1_fused_kernel(
    const _Float16* __restrict__ sf16, const _Float16* __restrict__ W1T,
    const float* __restrict__ b1, const float* __restrict__ ln_g,
    const float* __restrict__ ln_b, const float* __restrict__ W2,
    const int* __restrict__ idx, const float* __restrict__ q_pos,
    float* __restrict__ spos) {
  __shared__ float s_h[2][16 * C_];
  __shared__ float s_mu[2][16], s_rs[2][16];

  const int w = threadIdx.x >> 5, lane = threadIdx.x & 31;
  const int half = lane >> 4, r = lane & 15;
  const int rb = (blockIdx.x * 2 + w) * 16;

  for (int cg = 0; cg < C_ / 64; ++cg) {
    v8f acc0 = {}, acc1 = {}, acc2 = {}, acc3 = {};
    for (int kb = 0; kb < C_; kb += 32) {
      v16h a  = load_a16(sf16, C_, rb + r, kb, half);
      v16h b0 = load_bT(W1T, kb, cg * 64 + 0 * 16 + r, half);
      v16h b1f = load_bT(W1T, kb, cg * 64 + 1 * 16 + r, half);
      v16h b2 = load_bT(W1T, kb, cg * 64 + 2 * 16 + r, half);
      v16h b3 = load_bT(W1T, kb, cg * 64 + 3 * 16 + r, half);
      acc0 = __builtin_amdgcn_wmma_f32_16x16x32_f16(false, a, false, b0, (short)0, acc0, false, false);
      acc1 = __builtin_amdgcn_wmma_f32_16x16x32_f16(false, a, false, b1f, (short)0, acc1, false, false);
      acc2 = __builtin_amdgcn_wmma_f32_16x16x32_f16(false, a, false, b2, (short)0, acc2, false, false);
      acc3 = __builtin_amdgcn_wmma_f32_16x16x32_f16(false, a, false, b3, (short)0, acc3, false, false);
    }
    v8f* accs[4] = { &acc0, &acc1, &acc2, &acc3 };
#pragma unroll
    for (int j = 0; j < 4; ++j) {
      int col = cg * 64 + j * 16 + r;
      float bv = b1[col];
#pragma unroll
      for (int v = 0; v < 8; ++v)
        s_h[w][(8 * half + v) * C_ + col] = (*accs[j])[v] + bv;
    }
  }
  __syncthreads();

  if (lane < 16) {
    float mu = 0.f;
    for (int c = 0; c < C_; ++c) mu += s_h[w][lane * C_ + c];
    mu *= (1.f / C_);
    float var = 0.f;
    for (int c = 0; c < C_; ++c) { float d = s_h[w][lane * C_ + c] - mu; var += d * d; }
    var *= (1.f / C_);
    s_mu[w][lane] = mu;
    s_rs[w][lane] = rsqrtf(var + LN_EPS_);
  }
  __syncthreads();

  for (int c = lane; c < C_; c += 32) {
    float gg = ln_g[c], bb = ln_b[c];
#pragma unroll 4
    for (int row = 0; row < 16; ++row) {
      float x = (s_h[w][row * C_ + c] - s_mu[w][row]) * s_rs[w][row] * gg + bb;
      x = 0.5f * x * (1.f + erff(x * 0.70710678118654752f));   // exact gelu
      s_h[w][row * C_ + c] = x;
    }
  }
  __syncthreads();

  if (lane < 16) {
    int row = lane;
    int m = rb + row;
    int b = m / (G_ * N_ * K_);
    int n = (m / K_) % N_;
    int k = m % K_;
    int nb = idx[(b * N_ + n) * K_ + k];
#pragma unroll
    for (int j = 0; j < 3; ++j) {
      float o = 0.f;
      for (int c = 0; c < C_; ++c) o += s_h[w][row * C_ + c] * W2[c * 3 + j];
      spos[(size_t)m * 3 + j] = q_pos[(size_t)(b * N_ + nb) * 3 + j] + tanhf(o);
    }
  }
}

// ---------------------------------------------------------------------------
// helper: insert (d,i) into per-thread sorted top-3
// ---------------------------------------------------------------------------
DEVI void t3_insert(float d, int ii, float& d0, float& d1, float& d2, int& i0, int& i1, int& i2) {
  if (d < d2) {
    d2 = d; i2 = ii;
    if (d2 < d1) { float t = d1; d1 = d2; d2 = t; int ti = i1; i1 = i2; i2 = ti; }
    if (d1 < d0) { float t = d0; d0 = d1; d1 = t; int ti = i0; i0 = i1; i1 = ti; }
  }
}

// ---------------------------------------------------------------------------
// Kernel 6: wave per shifted point: top-3 NN over N, inverse-distance interp
// ---------------------------------------------------------------------------
__global__ __launch_bounds__(128) void interp_kernel(
    const float* __restrict__ spos, const float* __restrict__ q_pos,
    const float* __restrict__ q, _Float16* __restrict__ interp16) {
  int m    = (int)((blockIdx.x * (long long)blockDim.x + threadIdx.x) >> 5);
  int lane = threadIdx.x & 31;
  if (m >= M1_) return;
  int b = m / (G_ * N_ * K_);
  int g = (m / (N_ * K_)) % G_;
  int n = (m / K_) % N_;
  int k = m % K_;
  const float* base = q_pos + (size_t)b * N_ * 3;
  float px = spos[(size_t)m * 3 + 0], py = spos[(size_t)m * 3 + 1], pz = spos[(size_t)m * 3 + 2];
  float d0 = 3.4e38f, d1 = 3.4e38f, d2 = 3.4e38f;
  int   i0 = 0, i1 = 0, i2 = 0;
  for (int j = lane; j < N_; j += 32) {
    float dx = base[j * 3 + 0] - px, dy = base[j * 3 + 1] - py, dz = base[j * 3 + 2] - pz;
    t3_insert(dx * dx + dy * dy + dz * dz, j, d0, d1, d2, i0, i1, i2);
  }
#pragma unroll
  for (int off = 16; off > 0; off >>= 1) {
    float od0 = __shfl_xor(d0, off), od1 = __shfl_xor(d1, off), od2 = __shfl_xor(d2, off);
    int   oi0 = __shfl_xor(i0, off), oi1 = __shfl_xor(i1, off), oi2 = __shfl_xor(i2, off);
    t3_insert(od0, oi0, d0, d1, d2, i0, i1, i2);
    t3_insert(od1, oi1, d0, d1, d2, i0, i1, i2);
    t3_insert(od2, oi2, d0, d1, d2, i0, i1, i2);
  }
  float r0 = 1.f / (d0 + 1e-8f), r1 = 1.f / (d1 + 1e-8f), r2 = 1.f / (d2 + 1e-8f);
  float inv = 1.f / (r0 + r1 + r2);
  float w0 = r0 * inv, w1 = r1 * inv, w2 = r2 * inv;
  const float* qb = q + (size_t)b * N_ * C_ + g * GD_;
  _Float16* outp = interp16 + (size_t)((b * N_ + n) * K_ + k) * C_ + g * GD_;
  for (int d = lane; d < GD_; d += 32) {
    float v = w0 * qb[(size_t)i0 * C_ + d] + w1 * qb[(size_t)i1 * C_ + d] + w2 * qb[(size_t)i2 * C_ + d];
    outp[d] = (_Float16)v;
  }
}

// ---------------------------------------------------------------------------
// Kernel 7: fused attention. wave = one (b,n): kk/vv via WMMA into LDS,
//           logits, softmax over K=10, weighted vv sum.
// ---------------------------------------------------------------------------
__global__ __launch_bounds__(64) void attn_fused_kernel(
    const _Float16* __restrict__ interp16, const _Float16* __restrict__ WkT,
    const _Float16* __restrict__ WvT, const float* __restrict__ qp,
    _Float16* __restrict__ ao16) {
  __shared__ _Float16 s_kk[2][16 * C_];
  __shared__ _Float16 s_vv[2][16 * C_];
  __shared__ float    s_log[2][H_ * 16];
  __shared__ float    s_attn[2][H_ * 16];

  const int w = threadIdx.x >> 5, lane = threadIdx.x & 31;
  const int bn = blockIdx.x * 2 + w;            // < 4096
  const int half = lane >> 4, r = lane & 15;

  int rowg = bn * K_ + r;                       // rows 10..15 of tile are padding
  if (rowg > MI_ - 1) rowg = MI_ - 1;

  for (int cg = 0; cg < C_ / 64; ++cg) {
    v8f ak0 = {}, ak1 = {}, ak2 = {}, ak3 = {};
    v8f av0 = {}, av1 = {}, av2 = {}, av3 = {};
    for (int kb = 0; kb < C_; kb += 32) {
      v16h a   = load_a16(interp16, C_, rowg, kb, half);
      v16h bk0 = load_bT(WkT, kb, cg * 64 + 0 * 16 + r, half);
      v16h bk1 = load_bT(WkT, kb, cg * 64 + 1 * 16 + r, half);
      v16h bk2 = load_bT(WkT, kb, cg * 64 + 2 * 16 + r, half);
      v16h bk3 = load_bT(WkT, kb, cg * 64 + 3 * 16 + r, half);
      v16h bv0 = load_bT(WvT, kb, cg * 64 + 0 * 16 + r, half);
      v16h bv1 = load_bT(WvT, kb, cg * 64 + 1 * 16 + r, half);
      v16h bv2 = load_bT(WvT, kb, cg * 64 + 2 * 16 + r, half);
      v16h bv3 = load_bT(WvT, kb, cg * 64 + 3 * 16 + r, half);
      ak0 = __builtin_amdgcn_wmma_f32_16x16x32_f16(false, a, false, bk0, (short)0, ak0, false, false);
      ak1 = __builtin_amdgcn_wmma_f32_16x16x32_f16(false, a, false, bk1, (short)0, ak1, false, false);
      ak2 = __builtin_amdgcn_wmma_f32_16x16x32_f16(false, a, false, bk2, (short)0, ak2, false, false);
      ak3 = __builtin_amdgcn_wmma_f32_16x16x32_f16(false, a, false, bk3, (short)0, ak3, false, false);
      av0 = __builtin_amdgcn_wmma_f32_16x16x32_f16(false, a, false, bv0, (short)0, av0, false, false);
      av1 = __builtin_amdgcn_wmma_f32_16x16x32_f16(false, a, false, bv1, (short)0, av1, false, false);
      av2 = __builtin_amdgcn_wmma_f32_16x16x32_f16(false, a, false, bv2, (short)0, av2, false, false);
      av3 = __builtin_amdgcn_wmma_f32_16x16x32_f16(false, a, false, bv3, (short)0, av3, false, false);
    }
    v8f* aks[4] = { &ak0, &ak1, &ak2, &ak3 };
    v8f* avs[4] = { &av0, &av1, &av2, &av3 };
#pragma unroll
    for (int j = 0; j < 4; ++j) {
      int col = cg * 64 + j * 16 + r;
#pragma unroll
      for (int v = 0; v < 8; ++v) {
        int row = 8 * half + v;
        s_kk[w][row * C_ + col] = (_Float16)(*aks[j])[v];
        s_vv[w][row * C_ + col] = (_Float16)(*avs[j])[v];
      }
    }
  }
  __syncthreads();

  const float* qrow = qp + (size_t)bn * C_;
  for (int p = lane; p < H_ * K_; p += 32) {
    int h = p / K_, k = p % K_;
    float acc = 0.f;
    for (int d = 0; d < HD_; ++d)
      acc += qrow[h * HD_ + d] * (float)s_kk[w][k * C_ + h * HD_ + d];
    s_log[w][h * 16 + k] = acc * SCALE_;
  }
  __syncthreads();

  if (lane < H_) {
    int h = lane;
    float mx = -1e30f;
    for (int k = 0; k < K_; ++k) mx = fmaxf(mx, s_log[w][h * 16 + k]);
    float s = 0.f;
    for (int k = 0; k < K_; ++k) { float e = __expf(s_log[w][h * 16 + k] - mx); s_attn[w][h * 16 + k] = e; s += e; }
    float inv = 1.f / s;
    for (int k = 0; k < K_; ++k) s_attn[w][h * 16 + k] *= inv;
  }
  __syncthreads();

  for (int c = lane; c < C_; c += 32) {
    int h = c / HD_;
    float o = 0.f;
    for (int k = 0; k < K_; ++k)
      o += s_attn[w][h * 16 + k] * (float)s_vv[w][k * C_ + c];
    ao16[(size_t)bn * C_ + c] = (_Float16)o;
  }
}

// ---------------------------------------------------------------------------
// host side
// ---------------------------------------------------------------------------
extern "C" void kernel_launch(void* const* d_in, const int* in_sizes, int n_in,
                              void* d_out, int out_size, void* d_ws, size_t ws_size,
                              hipStream_t stream) {
  (void)in_sizes; (void)n_in; (void)out_size; (void)ws_size;
  const float* q     = (const float*)d_in[0];
  const float* q_pos = (const float*)d_in[1];
  const float* Wq    = (const float*)d_in[2];
  const float* Wk    = (const float*)d_in[3];
  const float* Wv    = (const float*)d_in[4];
  const float* Wvoff = (const float*)d_in[5];
  const float* W1    = (const float*)d_in[6];
  const float* b1    = (const float*)d_in[7];
  const float* ln_g  = (const float*)d_in[8];
  const float* ln_b  = (const float*)d_in[9];
  const float* W2    = (const float*)d_in[10];
  const float* Wproj = (const float*)d_in[11];
  const float* bproj = (const float*)d_in[12];
  float* out = (float*)d_out;

  char* ws = (char*)d_ws;
  size_t off = 0;
  auto take = [&](size_t bytes) -> void* {
    void* p = ws + off;
    off += (bytes + 255) & ~(size_t)255;
    return p;
  };
  _Float16* WqT    = (_Float16*)take((size_t)CC_ * 2);
  _Float16* WkT    = (_Float16*)take((size_t)CC_ * 2);
  _Float16* WvT    = (_Float16*)take((size_t)CC_ * 2);
  _Float16* WvoffT = (_Float16*)take((size_t)CC_ * 2);
  _Float16* W1T    = (_Float16*)take((size_t)CC_ * 2);
  _Float16* WprojT = (_Float16*)take((size_t)CC_ * 2);
  _Float16* q16    = (_Float16*)take((size_t)MQ_ * C_ * 2);
  int*      idx    = (int*)take((size_t)MQ_ * K_ * 4);
  float*    qp     = (float*)take((size_t)MQ_ * C_ * 4);
  float*    voff   = (float*)take((size_t)MQ_ * C_ * 4);
  _Float16* sf16   = (_Float16*)take((size_t)M1_ * C_ * 2);
  float*    spos   = (float*)take((size_t)M1_ * 3 * 4);
  _Float16* interp16 = (_Float16*)take((size_t)MI_ * C_ * 2);
  _Float16* ao16   = (_Float16*)take((size_t)MQ_ * C_ * 2);

  auto cvtT = [&](const float* s, _Float16* d) {
    cvt_f16_t_kernel<<<dim3((CC_ + 255) / 256), dim3(256), 0, stream>>>(s, d);
  };
  cvtT(Wq, WqT);
  cvtT(Wk, WkT);
  cvtT(Wv, WvT);
  cvtT(Wvoff, WvoffT);
  cvtT(W1, W1T);
  cvtT(Wproj, WprojT);
  cvt_f16_kernel<<<dim3((MQ_ * C_ + 255) / 256), dim3(256), 0, stream>>>(q, q16, MQ_ * C_);

  knn_kernel<<<dim3((MQ_ + 255) / 256), dim3(256), 0, stream>>>(q_pos, idx);

  // qp = q@Wq, voff = q@Wvoff : (M/16)*(C/64) waves, 4 waves/block
  {
    int waves = (MQ_ / 16) * (C_ / 64);      // 1536
    wmma_gemm_kernel<<<dim3(waves / 4), dim3(128), 0, stream>>>(q16, WqT, nullptr, qp, MQ_);
    wmma_gemm_kernel<<<dim3(waves / 4), dim3(128), 0, stream>>>(q16, WvoffT, nullptr, voff, MQ_);
  }

  {
    long long elems = (long long)M1_ * C_;
    build_sf_kernel<<<dim3((unsigned)((elems + 255) / 256)), dim3(256), 0, stream>>>(qp, voff, idx, sf16);
  }

  // fused W1 -> LN -> gelu -> W2 -> tanh -> shift_pos : 5120 waves, 2/block
  w1_fused_kernel<<<dim3(M1_ / 16 / 2), dim3(64), 0, stream>>>(
      sf16, W1T, b1, ln_g, ln_b, W2, idx, q_pos, spos);

  // top-3 + inverse distance interp : 81920 waves, 4/block
  interp_kernel<<<dim3(M1_ / 4), dim3(128), 0, stream>>>(spos, q_pos, q, interp16);

  // fused kk/vv + attention : 4096 waves, 2/block
  attn_fused_kernel<<<dim3(MQ_ / 2), dim3(64), 0, stream>>>(interp16, WkT, WvT, qp, ao16);

  // out = ao @ Wproj + bproj
  {
    int waves = (MQ_ / 16) * (C_ / 64);
    wmma_gemm_kernel<<<dim3(waves / 4), dim3(128), 0, stream>>>(ao16, WprojT, bproj, out, MQ_);
  }
}
